// SlotLSTMDecoder_66623532696258
// MI455X (gfx1250) — compile-verified, matrix operand order checked by reading, not run on hardware
//
#include <hip/hip_runtime.h>
#include <hip/hip_bf16.h>

// ---------------------------------------------------------------------------
// SlotLSTMDecoder for MI455X (gfx1250, wave32, WMMA)
//
// Phase 1: x_proj = x @ W_x^T + bias  via v_wmma_f32_16x16x32_f16 tiles,
//          K-loop ping-pong double-buffered so B-tile L2 latency overlaps
//          the matrix pipe instead of serializing (s_wait_loadcnt 0 per wmma).
// Phase 2: sequential LSTM scan, W_hh / W_out held in VGPRs as WMMA
//          A-fragments, W_slot@slot_emb precomputed to a lookup table.
// ---------------------------------------------------------------------------

#define NSLOT   128            // NUM_SLOTS (hidden)
#define NGATE   512            // 4 * NUM_SLOTS
#define INFEAT  1536           // INPUT_FEATS
#define NINT    41             // NUM_INTENTS
#define SEMB    32             // SLOT_EMBED
#define LSTMIN  1609           // NINT + INFEAT + SEMB
#define SEQ     16384

typedef __attribute__((ext_vector_type(16))) _Float16 v16h;
typedef __attribute__((ext_vector_type(8)))  float    v8f;

// ---- workspace layout (bytes) ----
#define OFF_XP      ((size_t)0)                       // 16384*512*4 = 33,554,432
#define OFF_WX16    ((size_t)33554432)                // 512*1536*2  =  1,572,864
#define OFF_WHH16   ((size_t)35127296)                // 512*128*2   =    131,072
#define OFF_WOUT16  ((size_t)35258368)                // 128*128*2   =     32,768
#define OFF_SLOTT   ((size_t)35291136)                // 128*512*4   =    262,144
#define OFF_INITC   ((size_t)35553280)                // 512*4
#define OFF_BIAS    ((size_t)35555328)                // 512*4

// B fragment (16 cols x 32 K, f16): lane (n + 16h) holds W row n, K chunks
// [k0+8h,+8) and [k0+16+8h,+8).  p points at row start + k0 + 8h.
__device__ __forceinline__ v16h load_bfrag(const _Float16* __restrict__ p)
{
    v16h b;
    #pragma unroll
    for (int i = 0; i < 8; ++i) { b[i] = p[i]; b[8 + i] = p[16 + i]; }
    return b;
}

// A fragment (16 rows x 32 K) from f32 source, converted to f16 in-register.
__device__ __forceinline__ v16h load_afrag_f32(const float* __restrict__ p)
{
    v16h a;
    #pragma unroll
    for (int i = 0; i < 8; ++i) {
        a[i]     = (_Float16)p[i];
        a[8 + i] = (_Float16)p[16 + i];
    }
    return a;
}

// ---------------------------------------------------------------------------
// Prep: convert W_x / W_hh / W_out to f16 with friendly layout
// ---------------------------------------------------------------------------
__global__ void k_convert(const float* __restrict__ W_ih,
                          const float* __restrict__ W_hh,
                          const float* __restrict__ W_out,
                          _Float16* __restrict__ wx16,
                          _Float16* __restrict__ whh16,
                          _Float16* __restrict__ wout16)
{
    int idx = blockIdx.x * 256 + threadIdx.x;
    const int NWX = NGATE * INFEAT;   // 786432
    const int NWH = NGATE * NSLOT;    // 65536
    const int NWO = NSLOT * NSLOT;    // 16384
    if (idx < NWX) {
        int j = idx / INFEAT, k = idx - j * INFEAT;
        wx16[idx] = (_Float16)W_ih[(size_t)j * LSTMIN + (NINT + SEMB) + k];
    } else if (idx < NWX + NWH) {
        int i2 = idx - NWX;
        whh16[i2] = (_Float16)W_hh[i2];
    } else if (idx < NWX + NWH + NWO) {
        int i2 = idx - NWX - NWH;
        wout16[i2] = (_Float16)W_out[i2];
    }
}

// ---------------------------------------------------------------------------
// Prep: slot_proj table (W_slot @ slot_emb^T, transposed to [slot][512]),
//       init column (W_slot @ init_state), fused bias (b_ih+b_hh+W_int col).
// ---------------------------------------------------------------------------
__global__ void k_slotproj(const float* __restrict__ W_ih,
                           const float* __restrict__ slot_emb,
                           const float* __restrict__ init_state,
                           const float* __restrict__ b_ih,
                           const float* __restrict__ b_hh,
                           const int*   __restrict__ intent,
                           float* __restrict__ slotT,
                           float* __restrict__ initcol,
                           float* __restrict__ bias512)
{
    int idx = blockIdx.x * 256 + threadIdx.x;
    if (idx < NSLOT * NGATE) {
        int s = idx / NGATE, j = idx - s * NGATE;
        const float* wrow = W_ih + (size_t)j * LSTMIN + NINT;   // W_slot row j
        const float* se   = slot_emb + s * SEMB;
        float acc = 0.f;
        #pragma unroll
        for (int e = 0; e < SEMB; ++e) acc += wrow[e] * se[e];
        slotT[idx] = acc;
    } else if (idx < NSLOT * NGATE + NGATE) {
        int j = idx - NSLOT * NGATE;
        const float* wrow = W_ih + (size_t)j * LSTMIN + NINT;
        float acc = 0.f;
        #pragma unroll
        for (int e = 0; e < SEMB; ++e) acc += wrow[e] * init_state[e];
        initcol[j] = acc;
        bias512[j] = b_ih[j] + b_hh[j] + W_ih[(size_t)j * LSTMIN + intent[0]];
    }
}

// ---------------------------------------------------------------------------
// Phase 1: x_proj[t][j] = sum_k x[t][k]*Wx[j][k] + bias[j]
// Block: 256 thr = 8 waves. Wave (wm,wn): rows [b*32+16*wm,+16), cols [128*wn,+128)
// 8 accumulators/wave, 48 K-steps, ping-pong double buffered (unroll by 2):
// loads for K-step k+1 are issued before the 8 WMMAs of K-step k.
// ---------------------------------------------------------------------------
__global__ __launch_bounds__(256)
void gemm_xproj(const float* __restrict__ x,
                const _Float16* __restrict__ wx16,
                const float* __restrict__ bias512,
                const int* __restrict__ d_end,
                float* __restrict__ xp)
{
    const int seq = (*d_end == -1) ? SEQ : *d_end;
    const int m0  = blockIdx.x * 32;
    if (m0 >= seq) return;

    const int tid  = threadIdx.x;
    const int wave = tid >> 5, lane = tid & 31;
    const int lm = lane & 15, hf = lane >> 4;
    const int wm = wave >> 2, wn = wave & 3;
    const int r0 = m0 + wm * 16;
    const int n0 = wn * 128;

    v8f acc[8];
    #pragma unroll
    for (int nt = 0; nt < 8; ++nt) {
        float bv = bias512[n0 + nt * 16 + lm];
        #pragma unroll
        for (int r = 0; r < 8; ++r) acc[nt][r] = bv;
    }

    const float*    xbase = x + (size_t)(r0 + lm) * INFEAT + 8 * hf;
    const _Float16* wbase = wx16 + (size_t)(n0 + lm) * INFEAT + 8 * hf;

    v16h a0, a1, b0[8], b1[8];

    // prologue: K-step 0 into buffer 0
    a0 = load_afrag_f32(xbase);
    #pragma unroll
    for (int nt = 0; nt < 8; ++nt)
        b0[nt] = load_bfrag(wbase + (size_t)nt * 16 * INFEAT);

    for (int kt = 0; kt < INFEAT / 32; kt += 2) {
        const int k1 = (kt + 1) * 32;
        // issue loads for K-step kt+1 (buffer 1) before consuming buffer 0
        a1 = load_afrag_f32(xbase + k1);
        #pragma unroll
        for (int nt = 0; nt < 8; ++nt)
            b1[nt] = load_bfrag(wbase + (size_t)nt * 16 * INFEAT + k1);

        #pragma unroll
        for (int nt = 0; nt < 8; ++nt)
            acc[nt] = __builtin_amdgcn_wmma_f32_16x16x32_f16(
                false, a0, false, b0[nt], (short)0, acc[nt], false, false);

        const int k2 = (kt + 2) * 32;
        if (kt + 2 < INFEAT / 32) {
            // issue loads for K-step kt+2 (buffer 0) before consuming buffer 1
            a0 = load_afrag_f32(xbase + k2);
            #pragma unroll
            for (int nt = 0; nt < 8; ++nt)
                b0[nt] = load_bfrag(wbase + (size_t)nt * 16 * INFEAT + k2);
        }

        #pragma unroll
        for (int nt = 0; nt < 8; ++nt)
            acc[nt] = __builtin_amdgcn_wmma_f32_16x16x32_f16(
                false, a1, false, b1[nt], (short)0, acc[nt], false, false);
    }

    // epilogue: uniform fast path (block fully in range), guarded tail otherwise
    if (m0 + 32 <= seq) {
        #pragma unroll
        for (int nt = 0; nt < 8; ++nt) {
            #pragma unroll
            for (int r = 0; r < 8; ++r)
                xp[(size_t)(r0 + r + 8 * hf) * NGATE + n0 + nt * 16 + lm] = acc[nt][r];
        }
    } else {
        #pragma unroll
        for (int nt = 0; nt < 8; ++nt) {
            #pragma unroll
            for (int r = 0; r < 8; ++r) {
                int row = r0 + r + 8 * hf;
                if (row < seq)
                    xp[(size_t)row * NGATE + n0 + nt * 16 + lm] = acc[nt][r];
            }
        }
    }
}

// ---------------------------------------------------------------------------
// Phase 2: persistent single-block sequential scan.
// 8 waves: W_hh as 4x4 resident A-fragments/wave (rows wave*64..+64),
// W_out as 4 resident A-fragments/wave (rows wave*16..+16).
// h broadcast into B-fragments (all 16 N-columns identical).
// ---------------------------------------------------------------------------
__global__ __launch_bounds__(256, 1)
void lstm_scan(const float* __restrict__ xp,
               const _Float16* __restrict__ whh16,
               const _Float16* __restrict__ wout16,
               const float* __restrict__ slotT,
               const float* __restrict__ initcol,
               const int* __restrict__ d_end,
               float* __restrict__ out)
{
    __shared__ float    gbias[NGATE];
    __shared__ float    gates[NGATE];
    __shared__ _Float16 h16[NSLOT];
    __shared__ float    pred[NSLOT];
    __shared__ int      sIdx;

    const int seq  = (*d_end == -1) ? SEQ : *d_end;
    const int tid  = threadIdx.x;
    const int wave = tid >> 5, lane = tid & 31;
    const int lm = lane & 15, hf = lane >> 4;

    // --- resident W_hh A-fragments: wave covers rows [wave*64, +64) ---
    v16h whhF[4][4];
    #pragma unroll
    for (int mt = 0; mt < 4; ++mt) {
        const int row = (wave * 4 + mt) * 16 + lm;
        #pragma unroll
        for (int kt = 0; kt < 4; ++kt)
            whhF[mt][kt] = load_bfrag(whh16 + (size_t)row * NSLOT + kt * 32 + 8 * hf);
    }
    // --- resident W_out A-fragments: wave covers rows [wave*16, +16) ---
    v16h woutF[4];
    {
        const int row = wave * 16 + lm;
        #pragma unroll
        for (int kt = 0; kt < 4; ++kt)
            woutF[kt] = load_bfrag(wout16 + (size_t)row * NSLOT + kt * 32 + 8 * hf);
    }

    float c = 0.f;                       // cell state, private to threads 0..127
    if (tid < NSLOT) h16[tid] = (_Float16)0.f;
    if (tid == 0) sIdx = 0;

    v16h bfrag[4];                       // h as B-fragment (h0 == 0)
    #pragma unroll
    for (int kt = 0; kt < 4; ++kt) {
        #pragma unroll
        for (int i = 0; i < 16; ++i) bfrag[kt][i] = (_Float16)0.f;
    }
    __syncthreads();

    for (int t = 0; t < seq; ++t) {
        // 1) per-step bias: x_proj[t] + slot_proj column (or init column at t=0)
        const float* xrow = xp + (size_t)t * NGATE;
        const float* scol = (t == 0) ? initcol : (slotT + (size_t)sIdx * NGATE);
        gbias[tid]       = xrow[tid]       + scol[tid];
        gbias[tid + 256] = xrow[tid + 256] + scol[tid + 256];
        if (t + 8 < seq)
            __builtin_prefetch(xp + (size_t)(t + 8) * NGATE + tid * 2, 0, 1);
        __syncthreads();

        // 2) gates = bias + W_hh @ h_prev   (WMMA from registers)
        {
            const int m0w = wave * 64;
            #pragma unroll
            for (int mt = 0; mt < 4; ++mt) {
                const int mbase = m0w + mt * 16;
                v8f acc;
                #pragma unroll
                for (int r = 0; r < 8; ++r) acc[r] = gbias[mbase + r + 8 * hf];
                #pragma unroll
                for (int kt = 0; kt < 4; ++kt)
                    acc = __builtin_amdgcn_wmma_f32_16x16x32_f16(
                        false, whhF[mt][kt], false, bfrag[kt], (short)0, acc, false, false);
                if (lm == 0) {
                    #pragma unroll
                    for (int r = 0; r < 8; ++r) gates[mbase + r + 8 * hf] = acc[r];
                }
            }
        }
        __syncthreads();

        // 3) LSTM cell update (threads 0..127 own one hidden element)
        if (tid < NSLOT) {
            float ig = 1.f / (1.f + __expf(-gates[tid]));
            float fg = 1.f / (1.f + __expf(-gates[NSLOT + tid]));
            float gg = tanhf(gates[2 * NSLOT + tid]);
            float og = 1.f / (1.f + __expf(-gates[3 * NSLOT + tid]));
            c = fg * c + ig * gg;
            float h = og * tanhf(c);
            h16[tid] = (_Float16)h;
        }
        __syncthreads();

        // rebuild h B-fragment from new h (serves pred now and W_hh next step)
        #pragma unroll
        for (int kt = 0; kt < 4; ++kt)
            bfrag[kt] = load_bfrag((const _Float16*)h16 + kt * 32 + 8 * hf);

        // 4) pred = W_out @ h  (wave w -> rows [16w,+16))
        {
            v8f acc;
            #pragma unroll
            for (int r = 0; r < 8; ++r) acc[r] = 0.f;
            #pragma unroll
            for (int kt = 0; kt < 4; ++kt)
                acc = __builtin_amdgcn_wmma_f32_16x16x32_f16(
                    false, woutF[kt], false, bfrag[kt], (short)0, acc, false, false);
            if (lm == 0) {
                #pragma unroll
                for (int r = 0; r < 8; ++r) {
                    const int row = wave * 16 + r + 8 * hf;
                    pred[row] = acc[r];
                    out[(size_t)t * NSLOT + row] = acc[r];
                }
            }
        }
        __syncthreads();

        // 5) argmax(pred) by wave 0 (first-index tie-break, like jnp.argmax)
        if (wave == 0) {
            float bv = pred[lane]; int bi = lane;
            #pragma unroll
            for (int q = 1; q < 4; ++q) {
                const int i2 = lane + q * 32;
                const float v = pred[i2];
                if (v > bv || (v == bv && i2 < bi)) { bv = v; bi = i2; }
            }
            #pragma unroll
            for (int off = 16; off > 0; off >>= 1) {
                const float ov = __shfl_xor(bv, off, 32);
                const int   oi = __shfl_xor(bi, off, 32);
                if (ov > bv || (ov == bv && oi < bi)) { bv = ov; bi = oi; }
            }
            if (lane == 0) sIdx = bi;
        }
        __syncthreads();
    }
}

// ---------------------------------------------------------------------------
extern "C" void kernel_launch(void* const* d_in, const int* in_sizes, int n_in,
                              void* d_out, int out_size, void* d_ws, size_t ws_size,
                              hipStream_t stream)
{
    const float* enc        = (const float*)d_in[0];
    const float* W_ih       = (const float*)d_in[1];
    const float* W_hh       = (const float*)d_in[2];
    const float* b_ih       = (const float*)d_in[3];
    const float* b_hh       = (const float*)d_in[4];
    const float* W_out      = (const float*)d_in[5];
    const float* slot_emb   = (const float*)d_in[6];
    const float* init_state = (const float*)d_in[7];
    const int*   intent     = (const int*)d_in[8];
    const int*   endidx     = (const int*)d_in[9];

    char* ws = (char*)d_ws;
    float*    xp      = (float*)   (ws + OFF_XP);
    _Float16* wx16    = (_Float16*)(ws + OFF_WX16);
    _Float16* whh16   = (_Float16*)(ws + OFF_WHH16);
    _Float16* wout16  = (_Float16*)(ws + OFF_WOUT16);
    float*    slotT   = (float*)   (ws + OFF_SLOTT);
    float*    initcol = (float*)   (ws + OFF_INITC);
    float*    bias512 = (float*)   (ws + OFF_BIAS);

    k_convert<<<3392, 256, 0, stream>>>(W_ih, W_hh, W_out, wx16, whh16, wout16);
    k_slotproj<<<258, 256, 0, stream>>>(W_ih, slot_emb, init_state, b_ih, b_hh,
                                        intent, slotT, initcol, bias512);
    gemm_xproj<<<SEQ / 32, 256, 0, stream>>>(enc, wx16, bias512, endidx, xp);
    lstm_scan<<<1, 256, 0, stream>>>(xp, whh16, wout16, slotT, initcol, endidx,
                                     (float*)d_out);
}
